// CNNTeacherModel_14551349198856
// MI455X (gfx1250) — matrix operand order repaired and checked by reading, not compiled, not run on hardware
//
#include <hip/hip_runtime.h>

#define B_ROWS   8192
#define H_DIM    1024
#define L_DIM    1024
#define N_HEADS  5
#define M_TILE   64                                   // rows per block (bucket pad unit)
#define NUM_M_TILES (B_ROWS / M_TILE + N_HEADS)       // 133
#define PAD_TOTAL (NUM_M_TILES * M_TILE)              // 8512
#define KCHUNK   32
#define NBLK     128                                  // N cols per block (2 wave-cols x 64)
#define A_STRIDE 40                                   // LDS row stride (80B, 16B aligned)

typedef __attribute__((ext_vector_type(16))) __bf16 v16bf;
typedef __attribute__((ext_vector_type(8)))  __bf16 v8bf;
typedef __attribute__((ext_vector_type(8)))  float  v8f;
typedef __attribute__((ext_vector_type(4)))  unsigned int v4u;

// ---- bf16 split helpers (RNE via bit math) ----
__device__ __forceinline__ unsigned short f2bf_bits(float x) {
    unsigned u = __float_as_uint(x);
    u += 0x7FFFu + ((u >> 16) & 1u);
    return (unsigned short)(u >> 16);
}
__device__ __forceinline__ float bfbits2f(unsigned short s) {
    return __uint_as_float(((unsigned)s) << 16);
}
// 16 contiguous bf16 -> v16bf fragment (works for LDS and global after inlining)
__device__ __forceinline__ v16bf load_frag32(const unsigned short* p) {
    v8bf l = __builtin_bit_cast(v8bf, *(const v4u*)p);
    v8bf h = __builtin_bit_cast(v8bf, *(const v4u*)(p + 8));
    return __builtin_shufflevector(l, h, 0,1,2,3,4,5,6,7,8,9,10,11,12,13,14,15);
}

// ---------------------------------------------------------------------------
// Kernel 0: one-time f32 -> (bf16 hi, bf16 lo) split conversion
// ---------------------------------------------------------------------------
__global__ void split_convert_kernel(const float* __restrict__ src,
                                     unsigned short* __restrict__ hi,
                                     unsigned short* __restrict__ lo,
                                     int n4) {
    int i = blockIdx.x * blockDim.x + threadIdx.x;
    if (i >= n4) return;
    float4 v = ((const float4*)src)[i];
    unsigned short hx = f2bf_bits(v.x), hy = f2bf_bits(v.y);
    unsigned short hz = f2bf_bits(v.z), hw = f2bf_bits(v.w);
    unsigned short lx = f2bf_bits(v.x - bfbits2f(hx));
    unsigned short ly = f2bf_bits(v.y - bfbits2f(hy));
    unsigned short lz = f2bf_bits(v.z - bfbits2f(hz));
    unsigned short lw = f2bf_bits(v.w - bfbits2f(hw));
    ((ushort4*)hi)[i] = make_ushort4(hx, hy, hz, hw);
    ((ushort4*)lo)[i] = make_ushort4(lx, ly, lz, lw);
}

// ---------------------------------------------------------------------------
// Kernel 1: counting-sort rows by head (groups 0..4), pad buckets to 64 rows.
// ---------------------------------------------------------------------------
__global__ void bucket_sort_kernel(const int* __restrict__ group,
                                   int* __restrict__ perm,
                                   int* __restrict__ tileHead) {
    __shared__ int cnt[N_HEADS];
    __shared__ int wr[N_HEADS];
    const int t = threadIdx.x;
    if (t < N_HEADS) cnt[t] = 0;
    __syncthreads();
    for (int i = t; i < B_ROWS; i += blockDim.x) {
        int g = group[i];
        if (g < N_HEADS) atomicAdd(&cnt[g], 1);
    }
    __syncthreads();
    if (t == 0) {
        int off = 0;
        for (int h = 0; h < N_HEADS; ++h) {
            wr[h] = off;
            int padded = ((cnt[h] + M_TILE - 1) / M_TILE) * M_TILE;
            int t0 = off / M_TILE, t1 = (off + padded) / M_TILE;
            for (int tt = t0; tt < t1; ++tt) tileHead[tt] = h;
            off += padded;
        }
        for (int tt = off / M_TILE; tt < NUM_M_TILES; ++tt) tileHead[tt] = -1;
    }
    __syncthreads();
    for (int i = t; i < PAD_TOTAL; i += blockDim.x) perm[i] = -1;
    __syncthreads();
    for (int i = t; i < B_ROWS; i += blockDim.x) {
        int g = group[i];
        if (g < N_HEADS) {
            int pos = atomicAdd(&wr[g], 1);
            perm[pos] = i;
        }
    }
}

// ---------------------------------------------------------------------------
// Kernel 2: rows with group==5 -> broadcast label value
// ---------------------------------------------------------------------------
__global__ void fill_label_rows(const int* __restrict__ group,
                                const int* __restrict__ labels,
                                float* __restrict__ out) {
    const int b = blockIdx.x;
    if (group[b] != N_HEADS) return;
    const float v = (float)labels[b];
    for (int j = threadIdx.x; j < L_DIM; j += blockDim.x)
        out[(size_t)b * L_DIM + j] = v;
}

// ---------------------------------------------------------------------------
// Kernel 3: gathered-A GEMM, 64x128 block tile, pre-split bf16 operands.
// 8 waves = 4(M) x 2(N); each wave: 1 A frag pair, 4 accumulators,
// 12 x v_wmma_f32_16x16x32_bf16 per 32-wide k-chunk.
// ---------------------------------------------------------------------------
__global__ __launch_bounds__(256)
void head_gemm_kernel(const unsigned short* __restrict__ Xhi,  // [B,H] bf16
                      const unsigned short* __restrict__ Xlo,
                      const unsigned short* __restrict__ Whi,  // [5,L,H] bf16
                      const unsigned short* __restrict__ Wlo,
                      const float* __restrict__ bias,          // [5,L]
                      const int*   __restrict__ perm,
                      const int*   __restrict__ tileHead,
                      float*       __restrict__ out)           // [B,L]
{
    __shared__ unsigned short AsHi[M_TILE][A_STRIDE];   // 5120 B
    __shared__ unsigned short AsLo[M_TILE][A_STRIDE];
    __shared__ int rows[M_TILE];
    __shared__ int headS;

    const int t = threadIdx.x;
    const int tileId = blockIdx.y;
    if (t == 0) headS = tileHead[tileId];
    if (t < M_TILE) rows[t] = perm[tileId * M_TILE + t];
    __syncthreads();
    const int head = headS;
    if (head < 0) return;  // block-uniform: fully padded tile

    const int wave = t >> 5;
    const int lane = t & 31;
    const int hlf  = lane >> 4;   // lane half
    const int col  = lane & 15;   // N within 16-tile (B/C/D layout)
    const int m    = lane & 15;   // M within 16-tile (A layout)
    const int mIdx = wave & 3;    // which 16-row A sub-tile
    const int nIdx = wave >> 2;   // which 64-col half
    const int nBase = blockIdx.x * NBLK + nIdx * 64;

    // B row pointers for this wave's 4 N-tiles (each lane: 16 contiguous K bf16)
    const size_t hBase = (size_t)head * L_DIM * H_DIM;
    const unsigned short* wh0 = Whi + hBase + (size_t)(nBase +  0 + col) * H_DIM + hlf * 16;
    const unsigned short* wh1 = Whi + hBase + (size_t)(nBase + 16 + col) * H_DIM + hlf * 16;
    const unsigned short* wh2 = Whi + hBase + (size_t)(nBase + 32 + col) * H_DIM + hlf * 16;
    const unsigned short* wh3 = Whi + hBase + (size_t)(nBase + 48 + col) * H_DIM + hlf * 16;
    const unsigned short* wl0 = Wlo + (wh0 - Whi);
    const unsigned short* wl1 = Wlo + (wh1 - Whi);
    const unsigned short* wl2 = Wlo + (wh2 - Whi);
    const unsigned short* wl3 = Wlo + (wh3 - Whi);

    // cooperative A staging: thread t loads one b128 (8 bf16) per matrix
    const int sRow = t >> 2;            // 0..63
    const int sSeg = (t & 3) * 8;       // 0,8,16,24
    const int rS = rows[sRow];
    const unsigned short* XhiS = Xhi + (size_t)(rS < 0 ? 0 : rS) * H_DIM + sSeg;
    const unsigned short* XloS = Xlo + (size_t)(rS < 0 ? 0 : rS) * H_DIM + sSeg;

    v8f acc0 = {}, acc1 = {}, acc2 = {}, acc3 = {};

    // preload chunk 0
    v4u ah = {}, al = {};
    if (rS >= 0) {
        ah = *(const v4u*)XhiS;
        al = *(const v4u*)XloS;
    }

    const unsigned short* aHiRow = &AsHi[mIdx * 16 + m][0];
    const unsigned short* aLoRow = &AsLo[mIdx * 16 + m][0];

    for (int k0 = 0; k0 < H_DIM; k0 += KCHUNK) {
        __syncthreads();                 // prior-iteration LDS reads done
        *(v4u*)&AsHi[sRow][sSeg] = ah;
        *(v4u*)&AsLo[sRow][sSeg] = al;
        __syncthreads();                 // tile visible

        // issue next chunk's A loads early (overlaps WMMA below)
        const int k1 = k0 + KCHUNK;
        if (k1 < H_DIM && rS >= 0) {
            ah = *(const v4u*)(XhiS + k1);
            al = *(const v4u*)(XloS + k1);
        }

        // A fragments: 16 contiguous K per lane (k = 16*hlf + s)
        v8bf a0 = __builtin_bit_cast(v8bf, *(const v4u*)(aHiRow + hlf * 16));
        v8bf a1 = __builtin_bit_cast(v8bf, *(const v4u*)(aHiRow + hlf * 16 + 8));
        v16bf aHi = __builtin_shufflevector(a0, a1, 0,1,2,3,4,5,6,7,8,9,10,11,12,13,14,15);
        v8bf a2 = __builtin_bit_cast(v8bf, *(const v4u*)(aLoRow + hlf * 16));
        v8bf a3 = __builtin_bit_cast(v8bf, *(const v4u*)(aLoRow + hlf * 16 + 8));
        v16bf aLo = __builtin_shufflevector(a2, a3, 0,1,2,3,4,5,6,7,8,9,10,11,12,13,14,15);

        // N-tile 0
        {
            v16bf bHi = load_frag32(wh0 + k0);
            v16bf bLo = load_frag32(wl0 + k0);
            acc0 = __builtin_amdgcn_wmma_f32_16x16x32_bf16(false, aHi, false, bHi, (short)0, acc0, false, false);
            acc0 = __builtin_amdgcn_wmma_f32_16x16x32_bf16(false, aHi, false, bLo, (short)0, acc0, false, false);
            acc0 = __builtin_amdgcn_wmma_f32_16x16x32_bf16(false, aLo, false, bHi, (short)0, acc0, false, false);
        }
        // N-tile 1
        {
            v16bf bHi = load_frag32(wh1 + k0);
            v16bf bLo = load_frag32(wl1 + k0);
            acc1 = __builtin_amdgcn_wmma_f32_16x16x32_bf16(false, aHi, false, bHi, (short)0, acc1, false, false);
            acc1 = __builtin_amdgcn_wmma_f32_16x16x32_bf16(false, aHi, false, bLo, (short)0, acc1, false, false);
            acc1 = __builtin_amdgcn_wmma_f32_16x16x32_bf16(false, aLo, false, bHi, (short)0, acc1, false, false);
        }
        // N-tile 2
        {
            v16bf bHi = load_frag32(wh2 + k0);
            v16bf bLo = load_frag32(wl2 + k0);
            acc2 = __builtin_amdgcn_wmma_f32_16x16x32_bf16(false, aHi, false, bHi, (short)0, acc2, false, false);
            acc2 = __builtin_amdgcn_wmma_f32_16x16x32_bf16(false, aHi, false, bLo, (short)0, acc2, false, false);
            acc2 = __builtin_amdgcn_wmma_f32_16x16x32_bf16(false, aLo, false, bHi, (short)0, acc2, false, false);
        }
        // N-tile 3
        {
            v16bf bHi = load_frag32(wh3 + k0);
            v16bf bLo = load_frag32(wl3 + k0);
            acc3 = __builtin_amdgcn_wmma_f32_16x16x32_bf16(false, aHi, false, bHi, (short)0, acc3, false, false);
            acc3 = __builtin_amdgcn_wmma_f32_16x16x32_bf16(false, aHi, false, bLo, (short)0, acc3, false, false);
            acc3 = __builtin_amdgcn_wmma_f32_16x16x32_bf16(false, aLo, false, bHi, (short)0, acc3, false, false);
        }
    }

    // bias + scatter-store (C/D layout: VGPR r -> M = r + hlf*8, N = col)
    const float bv0 = bias[head * L_DIM + nBase +  0 + col];
    const float bv1 = bias[head * L_DIM + nBase + 16 + col];
    const float bv2 = bias[head * L_DIM + nBase + 32 + col];
    const float bv3 = bias[head * L_DIM + nBase + 48 + col];
#pragma unroll
    for (int r = 0; r < 8; ++r) {
        const int rowG = rows[mIdx * 16 + r + hlf * 8];
        if (rowG >= 0) {
            float* op = out + (size_t)rowG * L_DIM + nBase + col;
            op[0]  = acc0[r] + bv0;
            op[16] = acc1[r] + bv1;
            op[32] = acc2[r] + bv2;
            op[48] = acc3[r] + bv3;
        }
    }
}

extern "C" void kernel_launch(void* const* d_in, const int* in_sizes, int n_in,
                              void* d_out, int out_size, void* d_ws, size_t ws_size,
                              hipStream_t stream) {
    const float* X      = (const float*)d_in[0];
    const float* W      = (const float*)d_in[1];
    const float* bias   = (const float*)d_in[2];
    const int*   group  = (const int*)d_in[3];
    const int*   labels = (const int*)d_in[4];
    float* out = (float*)d_out;

    // workspace layout
    char* ws = (char*)d_ws;
    int* perm     = (int*)ws;
    int* tileHead = perm + PAD_TOTAL;
    size_t off = (((size_t)(PAD_TOTAL + NUM_M_TILES)) * sizeof(int) + 63) & ~(size_t)63;
    const size_t nW = (size_t)N_HEADS * L_DIM * H_DIM;   // 5,242,880
    const size_t nX = (size_t)B_ROWS * H_DIM;            // 8,388,608
    unsigned short* Whi = (unsigned short*)(ws + off);
    unsigned short* Wlo = Whi + nW;
    unsigned short* Xhi = Wlo + nW;
    unsigned short* Xlo = Xhi + nX;

    const int w4 = (int)(nW / 4), x4 = (int)(nX / 4);
    split_convert_kernel<<<(w4 + 255) / 256, 256, 0, stream>>>(W, Whi, Wlo, w4);
    split_convert_kernel<<<(x4 + 255) / 256, 256, 0, stream>>>(X, Xhi, Xlo, x4);
    bucket_sort_kernel<<<1, 256, 0, stream>>>(group, perm, tileHead);
    fill_label_rows<<<B_ROWS, 256, 0, stream>>>(group, labels, out);
    dim3 grid(L_DIM / NBLK, NUM_M_TILES);
    head_gemm_kernel<<<grid, 256, 0, stream>>>(Xhi, Xlo, Whi, Wlo, bias,
                                               perm, tileHead, out);
}
